// MoELayer_33586644254992
// MI455X (gfx1250) — compile-verified, matrix operand order checked by reading, not run on hardware
//
#include <hip/hip_runtime.h>
#include <hip/hip_bf16.h>

// Problem constants (from reference: B=4,S=2048,H=1024,I=2048,E=8,TOP_K=2)
#define HDIM 1024
#define IDIM 2048
#define NEXP 8
#define TTOK 8192
#define THN  (TTOK * HDIM)   // 8388608 output elements (+1 aux scalar)
#define EPSF 1e-5f
#define LDB  40              // padded LDS row stride in halves (80B) -> conflict-free frag reads

typedef _Float16 v16h __attribute__((ext_vector_type(16)));
typedef _Float16 v4h  __attribute__((ext_vector_type(4)));
typedef float    v8f  __attribute__((ext_vector_type(8)));

__device__ __forceinline__ float safe_clampf(float v, float m) {
  if (v != v) return 0.f;                 // NaN -> 0
  return fminf(fmaxf(v, -m), m);          // +-inf -> +-m, clip otherwise
}

__device__ __forceinline__ v8f vzero8() {
  v8f z;
#pragma unroll
  for (int j = 0; j < 8; ++j) z[j] = 0.f;
  return z;
}

// A/B fragment for v_wmma_f32_16x16x32_f16 from row-major f16 memory
// (ISA 7.12.2): lane = 16*hi + r holds row r; halves[0..7] = K(8*hi..),
// halves[8..15] = K(16+8*hi..).
__device__ __forceinline__ v16h frag_f16(const _Float16* __restrict__ base, size_t ld) {
  const int lane = threadIdx.x & 31;
  const _Float16* p = base + (size_t)(lane & 15) * ld + ((lane >> 4) << 3);
  v16h f;
#pragma unroll
  for (int j = 0; j < 8; ++j) {
    f[j]     = p[j];
    f[j + 8] = p[j + 16];
  }
  return f;
}

__device__ __forceinline__ v8f wmma_f16(v16h a, v16h b, v8f c) {
  return __builtin_amdgcn_wmma_f32_16x16x32_f16(false, a, false, b, (short)0, c,
                                                false, false);
}

// ---- zero init ------------------------------------------------------------
__global__ __launch_bounds__(256) void zero_kernel(float* __restrict__ out,
                                                   float* __restrict__ auxbuf) {
  size_t idx = (size_t)blockIdx.x * 256 + threadIdx.x;
  if (idx <= (size_t)THN) out[idx] = 0.f;
  if (idx < 32) auxbuf[idx] = 0.f;
}

// ---- f32 -> f16 pre-conversion (4 elements / thread, vectorized) ----------
__global__ __launch_bounds__(256) void cvt_kernel(const float* __restrict__ s,
                                                  _Float16* __restrict__ d, int n4) {
  const int i = blockIdx.x * 256 + threadIdx.x;
  if (i < n4) {
    const float4 v = ((const float4*)s)[i];
    v4h o;
    o[0] = (_Float16)v.x; o[1] = (_Float16)v.y;
    o[2] = (_Float16)v.z; o[3] = (_Float16)v.w;
    ((v4h*)d)[i] = o;
  }
}

__global__ __launch_bounds__(256) void cvt_clamp_kernel(const float* __restrict__ s,
                                                        _Float16* __restrict__ d, int n4) {
  const int i = blockIdx.x * 256 + threadIdx.x;
  if (i < n4) {
    const float4 v = ((const float4*)s)[i];
    v4h o;
    o[0] = (_Float16)safe_clampf(v.x, 1000.f);
    o[1] = (_Float16)safe_clampf(v.y, 1000.f);
    o[2] = (_Float16)safe_clampf(v.z, 1000.f);
    o[3] = (_Float16)safe_clampf(v.w, 1000.f);
    ((v4h*)d)[i] = o;
  }
}

// ---- router: one 256-thread block per token -------------------------------
// auxbuf layout: [0..7] sum of probs, [8..15] top2 counts, [16] sum lse^2,
//                [17] sum entropy
__global__ __launch_bounds__(256) void router_kernel(const float* __restrict__ x,
                                                     const float* __restrict__ Wr,
                                                     float* __restrict__ comb,
                                                     float* __restrict__ auxbuf) {
  const int t = blockIdx.x;
  const float* xp = x + (size_t)t * HDIM;

  float vals[4];
  float s = 0.f, ss = 0.f;
#pragma unroll
  for (int j = 0; j < 4; ++j) {
    float v = safe_clampf(xp[threadIdx.x + j * 256], 1000.f);
    vals[j] = v;
    s += v;
    ss += v * v;
  }
#pragma unroll
  for (int off = 16; off; off >>= 1) {
    s  += __shfl_xor(s,  off);
    ss += __shfl_xor(ss, off);
  }
  __shared__ float sA[8], sB[8];
  __shared__ float sL[8][8];
  const int wid = threadIdx.x >> 5;
  if ((threadIdx.x & 31) == 0) { sA[wid] = s; sB[wid] = ss; }
  __syncthreads();
  float tot = 0.f, tot2 = 0.f;
#pragma unroll
  for (int w = 0; w < 8; ++w) { tot += sA[w]; tot2 += sB[w]; }
  const float mu  = tot * (1.f / HDIM);
  const float var = tot2 * (1.f / HDIM) - mu * mu;
  const float rsig = rsqrtf(var + EPSF);

  float lg[NEXP];
#pragma unroll
  for (int e = 0; e < NEXP; ++e) lg[e] = 0.f;
#pragma unroll
  for (int j = 0; j < 4; ++j) {
    const int h = threadIdx.x + j * 256;
    const float hn = safe_clampf((vals[j] - mu) * rsig, 100.f);
#pragma unroll
    for (int e = 0; e < NEXP; ++e) lg[e] += hn * Wr[(size_t)e * HDIM + h];
  }
#pragma unroll
  for (int e = 0; e < NEXP; ++e) {
#pragma unroll
    for (int off = 16; off; off >>= 1) lg[e] += __shfl_xor(lg[e], off);
  }
  if ((threadIdx.x & 31) == 0) {
#pragma unroll
    for (int e = 0; e < NEXP; ++e) sL[wid][e] = lg[e];
  }
  __syncthreads();

  if (threadIdx.x == 0) {
    float logits[NEXP];
#pragma unroll
    for (int e = 0; e < NEXP; ++e) {
      float v = 0.f;
#pragma unroll
      for (int w = 0; w < 8; ++w) v += sL[w][e];
      logits[e] = fminf(fmaxf(v, -20.f), 20.f);
    }
    float mx = logits[0];
#pragma unroll
    for (int e = 1; e < NEXP; ++e) mx = fmaxf(mx, logits[e]);
    float p[NEXP], den = 0.f;
#pragma unroll
    for (int e = 0; e < NEXP; ++e) { p[e] = __expf(logits[e] - mx); den += p[e]; }
    const float lse = mx + logf(den);
    const float rden = 1.f / den;
#pragma unroll
    for (int e = 0; e < NEXP; ++e)
      p[e] = fminf(fmaxf(p[e] * rden, EPSF), 1.0f);   // probs == lp

    // top-2 (ties -> lower index, like jax.lax.top_k)
    int i0 = 0;
#pragma unroll
    for (int e = 1; e < NEXP; ++e) if (p[e] > p[i0]) i0 = e;
    int i1 = (i0 == 0) ? 1 : 0;
#pragma unroll
    for (int e = 0; e < NEXP; ++e)
      if (e != i0 && p[e] > p[i1]) i1 = e;
    const float vsum = fmaxf(p[i0] + p[i1], EPSF);
    const float w0 = p[i0] / vsum, w1 = p[i1] / vsum;

    float* cp = comb + (size_t)t * NEXP;
#pragma unroll
    for (int e = 0; e < NEXP; ++e) cp[e] = 0.f;
    cp[i0] = w0;
    cp[i1] = w1;

    // aux partials
    atomicAdd(&auxbuf[8 + i0], 1.f);
    atomicAdd(&auxbuf[8 + i1], 1.f);
    atomicAdd(&auxbuf[16], lse * lse);
    float ent = 0.f;
#pragma unroll
    for (int e = 0; e < NEXP; ++e) {
      const float ps = fminf(fmaxf(p[e], EPSF), 1.f - EPSF);
      ent -= ps * logf(ps);
      atomicAdd(&auxbuf[e], p[e]);
    }
    atomicAdd(&auxbuf[17], ent);
  }
}

// ---- fused gate/up GEMM with LDS-staged B tiles ---------------------------
// Block = 8 waves, tile = 128 tokens x 64 I-cols.  B tiles (Wg,Wu: 64x32 f16)
// double-buffered in LDS (padded stride LDB=40 halves -> conflict-free
// ds_load_b128 fragment reads); one barrier per K-step, global prefetch of
// the next tile overlaps the WMMA phase.
__global__ __launch_bounds__(256) void gateup_kernel(const _Float16* __restrict__ xh,
                                                     const _Float16* __restrict__ wgh,
                                                     const _Float16* __restrict__ wuh,
                                                     _Float16* __restrict__ act) {
  __shared__ _Float16 sB[2][2][64 * LDB];   // [buf][gate/up][row*LDB+k]  20KB

  const int nt = blockIdx.x & 31;           // IDIM/64 = 32 column tiles
  const int mt = blockIdx.x >> 5;           // TTOK/128 = 64 row tiles
  const int i0 = nt * 64;
  const int t0 = mt * 128;
  const int wv = threadIdx.x >> 5;
  const int lane = threadIdx.x & 31;
  const int tw = t0 + wv * 16;              // this wave's 16 token rows

  // cooperative staging map: row = tid>>2 (0..63), 8-half segment = tid&3
  const int srow = threadIdx.x >> 2;
  const int sseg = (threadIdx.x & 3) << 3;
  const size_t goff = (size_t)(i0 + srow) * HDIM + sseg;

  v8f gacc[4], uacc[4];
#pragma unroll
  for (int n = 0; n < 4; ++n) { gacc[n] = vzero8(); uacc[n] = vzero8(); }

  uint4 rg = *(const uint4*)(wgh + goff);   // prologue: tile k=0
  uint4 ru = *(const uint4*)(wuh + goff);

  const int br = lane & 15;
  const int bk = (lane >> 4) << 3;

  for (int kk = 0; kk < HDIM / 32; ++kk) {
    const int buf = kk & 1;
    *(uint4*)&sB[buf][0][srow * LDB + sseg] = rg;
    *(uint4*)&sB[buf][1][srow * LDB + sseg] = ru;
    __syncthreads();
    if (kk + 1 < HDIM / 32) {               // prefetch next tile (uniform branch)
      const size_t g2 = goff + (size_t)(kk + 1) * 32;
      rg = *(const uint4*)(wgh + g2);
      ru = *(const uint4*)(wuh + g2);
    }
    const v16h a = frag_f16(xh + (size_t)tw * HDIM + kk * 32, HDIM);
#pragma unroll
    for (int n = 0; n < 4; ++n) {
      v16h bg, bu;
      {
        const _Float16* p = &sB[buf][0][(16 * n + br) * LDB + bk];
#pragma unroll
        for (int j = 0; j < 8; ++j) { bg[j] = p[j]; bg[j + 8] = p[j + 16]; }
      }
      gacc[n] = wmma_f16(a, bg, gacc[n]);
      {
        const _Float16* p = &sB[buf][1][(16 * n + br) * LDB + bk];
#pragma unroll
        for (int j = 0; j < 8; ++j) { bu[j] = p[j]; bu[j + 8] = p[j + 16]; }
      }
      uacc[n] = wmma_f16(a, bu, uacc[n]);
    }
  }

  // epilogue: act = clamp(silu(g)) * clamp(u)
  const int hi = lane >> 4;
  const int nc = lane & 15;
#pragma unroll
  for (int n = 0; n < 4; ++n) {
#pragma unroll
    for (int v = 0; v < 8; ++v) {
      const int m = v + (hi << 3);
      float g = gacc[n][v];
      float u = uacc[n][v];
      g = safe_clampf(g / (1.f + __expf(-g)), 1000.f);
      u = safe_clampf(u, 1000.f);
      act[(size_t)(tw + m) * IDIM + (i0 + 16 * n + nc)] = (_Float16)(g * u);
    }
  }
}

// ---- down GEMM + weighted accumulate (LDS-staged Wd tiles) ----------------
// experts:  out += comb[t,e] * clamp(act@Wd^T, 1000)
// shared:   out += clamp(act@Wd^T * sigmoid(sgate), 1000)   (comb == nullptr)
__global__ __launch_bounds__(256) void down_kernel(const _Float16* __restrict__ act,
                                                   const _Float16* __restrict__ wdh,
                                                   const float* __restrict__ comb,
                                                   const float* __restrict__ sgate,
                                                   int e,
                                                   float* __restrict__ out) {
  __shared__ _Float16 sB[2][64 * LDB];      // 10KB

  const int nt = blockIdx.x & 15;           // HDIM/64 = 16 column tiles
  const int mt = blockIdx.x >> 4;           // TTOK/128 = 64 row tiles
  const int h0 = nt * 64;
  const int t0 = mt * 128;
  const int wv = threadIdx.x >> 5;
  const int lane = threadIdx.x & 31;
  const int tw = t0 + wv * 16;

  const int srow = threadIdx.x >> 2;
  const int sseg = (threadIdx.x & 3) << 3;
  const size_t goff = (size_t)(h0 + srow) * IDIM + sseg;

  v8f acc[4];
#pragma unroll
  for (int n = 0; n < 4; ++n) acc[n] = vzero8();

  uint4 rd = *(const uint4*)(wdh + goff);

  const int br = lane & 15;
  const int bk = (lane >> 4) << 3;

  for (int kk = 0; kk < IDIM / 32; ++kk) {
    const int buf = kk & 1;
    *(uint4*)&sB[buf][srow * LDB + sseg] = rd;
    __syncthreads();
    if (kk + 1 < IDIM / 32) {
      rd = *(const uint4*)(wdh + goff + (size_t)(kk + 1) * 32);
    }
    const v16h a = frag_f16(act + (size_t)tw * IDIM + kk * 32, IDIM);
#pragma unroll
    for (int n = 0; n < 4; ++n) {
      v16h b;
      const _Float16* p = &sB[buf][(16 * n + br) * LDB + bk];
#pragma unroll
      for (int j = 0; j < 8; ++j) { b[j] = p[j]; b[j + 8] = p[j + 16]; }
      acc[n] = wmma_f16(a, b, acc[n]);
    }
  }

  const int hi = lane >> 4;
  const int nc = lane & 15;
  const bool isShared = (comb == nullptr);
  float sgv = 1.f;
  if (isShared) sgv = 1.f / (1.f + __expf(-sgate[0]));

  float cw[8];
#pragma unroll
  for (int v = 0; v < 8; ++v) {
    cw[v] = isShared ? 0.f : comb[(size_t)(tw + v + (hi << 3)) * NEXP + e];
  }

#pragma unroll
  for (int n = 0; n < 4; ++n) {
#pragma unroll
    for (int v = 0; v < 8; ++v) {
      const int m = v + (hi << 3);
      float val = acc[n][v];
      if (isShared) val = safe_clampf(val * sgv, 1000.f);
      else          val = safe_clampf(val, 1000.f) * cw[v];
      const size_t o = (size_t)(tw + m) * HDIM + (h0 + 16 * n + nc);
      out[o] += val;   // tiles partition [T,H]; kernels are stream-ordered
    }
  }
}

// ---- final clamp + aux loss ----------------------------------------------
__global__ __launch_bounds__(256) void final_kernel(float* __restrict__ out,
                                                    const float* __restrict__ auxbuf) {
  const size_t idx = (size_t)blockIdx.x * 256 + threadIdx.x;
  if (idx < (size_t)THN) out[idx] = safe_clampf(out[idx], 1000.f);
  if (idx == 0) {
    const float Tf = (float)TTOK;
    float lb = 0.f, usage = 0.f;
#pragma unroll
    for (int e = 0; e < NEXP; ++e) {
      const float tpe = auxbuf[8 + e] / (Tf * 2.f);
      const float avg = auxbuf[e] / Tf;
      lb += tpe * avg;
      usage += (tpe > 0.01f) ? 1.f : 0.f;
    }
    lb *= (float)NEXP;
    const float z_loss = (auxbuf[16] / Tf) * 0.001f;
    const float ent_loss = fmaxf(0.f, logf((float)NEXP) - auxbuf[17] / Tf) * 0.01f;
    const float util_loss = (1.f - usage / (float)NEXP) * 0.1f;
    float aux = lb + z_loss + ent_loss + util_loss;
    if (!(aux == aux)) aux = 0.1f;
    aux = fminf(fmaxf(aux, 0.f), 10.f);
    if (fabsf(aux) > 1e30f) aux = 0.1f;
    out[THN] = aux;
  }
}

// ---- host-side launch ------------------------------------------------------
extern "C" void kernel_launch(void* const* d_in, const int* in_sizes, int n_in,
                              void* d_out, int out_size, void* d_ws, size_t ws_size,
                              hipStream_t stream) {
  (void)in_sizes; (void)n_in; (void)out_size; (void)ws_size;
  const float* x   = (const float*)d_in[0];   // [T,H]
  const float* Wr  = (const float*)d_in[1];   // [E,H]
  const float* Wg  = (const float*)d_in[2];   // [E,I,H]
  const float* Wu  = (const float*)d_in[3];   // [E,I,H]
  const float* Wd  = (const float*)d_in[4];   // [E,H,I]
  const float* Wsg = (const float*)d_in[5];   // [I,H]
  const float* Wsu = (const float*)d_in[6];   // [I,H]
  const float* Wsd = (const float*)d_in[7];   // [H,I]
  const float* sg  = (const float*)d_in[8];   // [1]
  float* out = (float*)d_out;

  // workspace layout (all 16B aligned)
  char* wsb = (char*)d_ws;
  float*    comb   = (float*)wsb;                                   // T*E f32  (256KB)
  float*    auxbuf = comb + (size_t)TTOK * NEXP;                    // 32 f32
  _Float16* act    = (_Float16*)(wsb + 262272);                     // [T,I] f16 (32MB)
  _Float16* xh     = (_Float16*)(wsb + 262272 + 33554432);          // [T,H] f16 (16MB)
  _Float16* wgh    = (_Float16*)(wsb + 262272 + 33554432 + 16777216);           // 4MB
  _Float16* wuh    = (_Float16*)(wsb + 262272 + 33554432 + 16777216 + 4194304); // 4MB
  _Float16* wdh    = (_Float16*)(wsb + 262272 + 33554432 + 16777216 + 8388608); // 4MB

  const dim3 blk(256);
  const int n4x = THN / 4;                 // 2097152
  const int n4w = IDIM * HDIM / 4;         // 524288

  zero_kernel<<<dim3((THN + 1 + 255) / 256), blk, 0, stream>>>(out, auxbuf);
  cvt_clamp_kernel<<<dim3(n4x / 256), blk, 0, stream>>>(x, xh, n4x);
  router_kernel<<<dim3(TTOK), blk, 0, stream>>>(x, Wr, comb, auxbuf);

  const int gu_blocks = (TTOK / 128) * (IDIM / 64);   // 2048
  const int dn_blocks = (TTOK / 128) * (HDIM / 64);   // 1024
  for (int e = 0; e < NEXP; ++e) {
    const size_t wo = (size_t)e * IDIM * HDIM;
    cvt_kernel<<<dim3(n4w / 256), blk, 0, stream>>>(Wg + wo, wgh, n4w);
    cvt_kernel<<<dim3(n4w / 256), blk, 0, stream>>>(Wu + wo, wuh, n4w);
    cvt_kernel<<<dim3(n4w / 256), blk, 0, stream>>>(Wd + wo, wdh, n4w);
    gateup_kernel<<<dim3(gu_blocks), blk, 0, stream>>>(xh, wgh, wuh, act);
    down_kernel<<<dim3(dn_blocks), blk, 0, stream>>>(act, wdh, comb, nullptr, e, out);
  }
  // shared expert
  cvt_kernel<<<dim3(n4w / 256), blk, 0, stream>>>(Wsg, wgh, n4w);
  cvt_kernel<<<dim3(n4w / 256), blk, 0, stream>>>(Wsu, wuh, n4w);
  cvt_kernel<<<dim3(n4w / 256), blk, 0, stream>>>(Wsd, wdh, n4w);
  gateup_kernel<<<dim3(gu_blocks), blk, 0, stream>>>(xh, wgh, wuh, act);
  down_kernel<<<dim3(dn_blocks), blk, 0, stream>>>(act, wdh, nullptr, sg, 0, out);

  final_kernel<<<dim3((THN + 255) / 256), blk, 0, stream>>>(out, auxbuf);
}
